// DFTMultiHeadAttention_63642825392589
// MI455X (gfx1250) — matrix operand order repaired and checked by reading, not compiled, not run on hardware
//
#include <hip/hip_runtime.h>
#include <hip/hip_bf16.h>
#include <stdint.h>

typedef __bf16 bf16_t;
typedef __attribute__((ext_vector_type(16))) __bf16 v16bf;
typedef __attribute__((ext_vector_type(8)))  __bf16 v8bf;
typedef __attribute__((ext_vector_type(8)))  float  v8f;

union V16 { v16bf v; v8bf h[2]; };

static constexpr int Bb = 2, Ss = 2048, Ee = 1024, Hh = 16, Dd = 64;
static constexpr int Mtot = Bb * Ss; // 4096

// ---------------- fp32 -> bf16 conversion ----------------
__global__ void cvt_f32_bf16(const float* __restrict__ src, bf16_t* __restrict__ dst, int n) {
    int i = blockIdx.x * blockDim.x + threadIdx.x;
    if (i < n) dst[i] = (bf16_t)src[i];
}

// ---------------- NT GEMM: Y[m,n] = sum_k A[m,k]*W[n,k] + bias[n] ----------------
// MODE 0: store bf16 to [B,H,S,D]   (Q, K for attention)
// MODE 1: store bf16 to [B,H,D,S]   (V transposed for attention)
// MODE 2: store fp32 to [M,N]       (final output projection)
template <int MODE>
__global__ __launch_bounds__(256)
void gemm_nt_bf16(const bf16_t* __restrict__ A, const bf16_t* __restrict__ W,
                  const float* __restrict__ bias, bf16_t* __restrict__ Yb,
                  float* __restrict__ Yf, int M, int N, int K) {
    const int lane = threadIdx.x & 31;
    const int wid  = threadIdx.x >> 5;
    const int wrow = wid & 3;          // 4 wave rows  -> 128 rows / block
    const int wcol = wid >> 2;         // 2 wave cols  -> 128 cols / block
    const int m0 = blockIdx.x * 128 + wrow * 32;
    const int n0 = blockIdx.y * 128 + wcol * 64;
    const int lm   = lane & 15;
    const int half = lane >> 4;
    const int klo  = half ? 16 : 0;    // A-frag K split (16x32 bf16 layout)
    const int kb   = half ? 16 : 0;    // B-frag K split (32x16 bf16 layout)

    v8f zc = {};
    v8f acc[2][4];
#pragma unroll
    for (int i = 0; i < 2; i++)
#pragma unroll
        for (int j = 0; j < 4; j++) acc[i][j] = zc;

    for (int k0 = 0; k0 < K; k0 += 32) {
        V16 a[2];
#pragma unroll
        for (int i = 0; i < 2; i++) {
            const bf16_t* ap = A + (size_t)(m0 + i * 16 + lm) * K + k0 + klo;
            a[i].h[0] = *(const v8bf*)(ap);
            a[i].h[1] = *(const v8bf*)(ap + 8);
        }
        V16 bw[4];
#pragma unroll
        for (int j = 0; j < 4; j++) {
            const bf16_t* wp = W + (size_t)(n0 + j * 16 + lm) * K + k0 + kb;
            bw[j].v = *(const v16bf*)(wp);
        }
#pragma unroll
        for (int i = 0; i < 2; i++)
#pragma unroll
            for (int j = 0; j < 4; j++)
                acc[i][j] = __builtin_amdgcn_wmma_f32_16x16x32_bf16(
                    false, a[i].v, false, bw[j].v, (short)0, acc[i][j], false, false);
    }

#pragma unroll
    for (int i = 0; i < 2; i++) {
#pragma unroll
        for (int j = 0; j < 4; j++) {
#pragma unroll
            for (int r = 0; r < 8; r++) {
                const int gm = m0 + i * 16 + r + half * 8;
                const int gn = n0 + j * 16 + lm;
                const float y = acc[i][j][r] + bias[gn];
                if (MODE == 2) {
                    Yf[(size_t)gm * N + gn] = y;
                } else {
                    const int b = gm >> 11, s = gm & (Ss - 1);
                    const int h = gn >> 6,  d = gn & (Dd - 1);
                    const size_t addr = (MODE == 0)
                        ? ((size_t)((b * Hh + h) * Ss + s)) * Dd + d
                        : ((size_t)((b * Hh + h) * Dd + d)) * Ss + s;
                    Yb[addr] = (bf16_t)y;
                }
            }
        }
    }
}

// ---- async global->LDS staging of a 64x(64 bf16) tile, rows padded to 72 elems ----
// Each of 128 threads copies 4 x 16B chunks; tracked on ASYNCcnt (CDNA5 path).
__device__ __forceinline__ void stage_tile_async(bf16_t* lds_tile, const bf16_t* gbase,
                                                 uint32_t goff_base, uint32_t grow_stride,
                                                 int tid) {
    const uint32_t lbase = (uint32_t)(uintptr_t)lds_tile;  // low 32 bits = LDS byte addr
#pragma unroll
    for (int it = 0; it < 4; it++) {
        const int c   = tid + it * 128;       // chunk id 0..511
        const int row = c >> 3;               // 8 chunks (128B) per row
        const int o16 = c & 7;
        const uint32_t lds  = lbase + (uint32_t)(row * 144 + o16 * 16);
        const uint32_t goff = goff_base + (uint32_t)row * grow_stride + (uint32_t)(o16 * 16);
        asm volatile("global_load_async_to_lds_b128 %0, %1, %2"
                     :
                     : "v"(lds), "v"(goff), "s"(gbase)
                     : "memory");
    }
}

__device__ __forceinline__ void wait_async_all() {
    asm volatile("s_wait_asynccnt 0x0" ::: "memory");
}

// ---------------- fused flash attention with logits = 64*|q.k| ----------------
// grid: (S/64, H, B), block: 128 threads (4 waves); wave owns 16 query rows.
// K/V tiles double-buffered in LDS via async copies; prefetch overlaps WMMA+softmax.
__global__ __launch_bounds__(128)
void flash_dft_attn(const bf16_t* __restrict__ Q, const bf16_t* __restrict__ Kh,
                    const bf16_t* __restrict__ Vt, bf16_t* __restrict__ O) {
    __shared__ bf16_t p_lds[4 * 16 * 72];   // per-wave 16x64 P tile
    __shared__ bf16_t k_lds[2][64 * 72];    // double-buffered K tile (rows j, cols d)
    __shared__ bf16_t v_lds[2][64 * 72];    // double-buffered V^T tile (rows d, cols j)

    const int tid  = threadIdx.x;
    const int lane = tid & 31;
    const int wid  = tid >> 5;
    const int lm   = lane & 15;
    const int half = lane >> 4;
    const int klo  = half ? 16 : 0;
    const int kb   = half ? 16 : 0;
    const int h = blockIdx.y, b = blockIdx.z;
    const int bh = b * Hh + h;
    const int q0 = blockIdx.x * 64 + wid * 16;

    const bf16_t* kbase = Kh + (size_t)bh * Ss * Dd;   // row j stride = 128B
    const bf16_t* vbase = Vt + (size_t)bh * Dd * Ss;   // row d stride = 4096B

    // Q A-fragments (2 K-steps over D=64), resident for whole kernel
    V16 aq[2];
    {
        const bf16_t* qp = Q + ((size_t)bh * Ss + q0 + lm) * Dd;
#pragma unroll
        for (int ks = 0; ks < 2; ks++) {
            aq[ks].h[0] = *(const v8bf*)(qp + ks * 32 + klo);
            aq[ks].h[1] = *(const v8bf*)(qp + ks * 32 + klo + 8);
        }
    }

    // prologue: stage first K/V tile
    stage_tile_async(k_lds[0], kbase, 0u, 128u, tid);
    stage_tile_async(v_lds[0], vbase, 0u, (uint32_t)(Ss * 2), tid);

    v8f zc = {};
    v8f oacc[4];
#pragma unroll
    for (int j = 0; j < 4; j++) oacc[j] = zc;
    float rmax[8], rsum[8];
#pragma unroll
    for (int r = 0; r < 8; r++) { rmax[r] = -3.0e38f; rsum[r] = 0.f; }

    bf16_t* myP = p_lds + wid * 16 * 72;

    for (int j0 = 0; j0 < Ss; j0 += 64) {
        const int buf = (j0 >> 6) & 1;
        wait_async_all();      // my staged copies (current buf) are in LDS
        __syncthreads();       // everyone's copies visible; prev-iter readers done

        // prefetch next tile into the other buffer (overlaps with compute below)
        if (j0 + 64 < Ss) {
            stage_tile_async(k_lds[buf ^ 1], kbase, (uint32_t)((j0 + 64) * 128), 128u, tid);
            stage_tile_async(v_lds[buf ^ 1], vbase, (uint32_t)((j0 + 64) * 2),
                             (uint32_t)(Ss * 2), tid);
        }
        const bf16_t* kt = k_lds[buf];
        const bf16_t* vt = v_lds[buf];

        // ---- scores tile: 16 x 64 (K B-frags from LDS) ----
        v8f sc[4];
#pragma unroll
        for (int jf = 0; jf < 4; jf++) sc[jf] = zc;
#pragma unroll
        for (int jf = 0; jf < 4; jf++) {
            const bf16_t* kp = kt + (jf * 16 + lm) * 72;
#pragma unroll
            for (int ks = 0; ks < 2; ks++) {
                V16 bk_;
                bk_.v = *(const v16bf*)(kp + ks * 32 + kb);
                sc[jf] = __builtin_amdgcn_wmma_f32_16x16x32_bf16(
                    false, aq[ks].v, false, bk_.v, (short)0, sc[jf], false, false);
            }
        }

        // ---- logits = D * |score| (Parseval collapse of DFT attention) ----
        float lg[4][8];
#pragma unroll
        for (int jf = 0; jf < 4; jf++)
#pragma unroll
            for (int r = 0; r < 8; r++) lg[jf][r] = 64.f * fabsf(sc[jf][r]);

        // ---- online softmax; C-layout rows reduce across 16-lane groups ----
        float p[4][8];
#pragma unroll
        for (int r = 0; r < 8; r++) {
            float m = fmaxf(fmaxf(lg[0][r], lg[1][r]), fmaxf(lg[2][r], lg[3][r]));
            m = fmaxf(m, __shfl_xor(m, 1));
            m = fmaxf(m, __shfl_xor(m, 2));
            m = fmaxf(m, __shfl_xor(m, 4));
            m = fmaxf(m, __shfl_xor(m, 8));
            const float nm    = fmaxf(rmax[r], m);
            const float scale = __expf(rmax[r] - nm);
            rmax[r] = nm;
            float s = 0.f;
#pragma unroll
            for (int jf = 0; jf < 4; jf++) {
                const float e = __expf(lg[jf][r] - nm);
                p[jf][r] = e;
                s += e;
            }
            s += __shfl_xor(s, 1);
            s += __shfl_xor(s, 2);
            s += __shfl_xor(s, 4);
            s += __shfl_xor(s, 8);
            rsum[r] = rsum[r] * scale + s;
#pragma unroll
            for (int df = 0; df < 4; df++) oacc[df][r] *= scale;
        }

        // ---- P: C-layout -> A-layout via LDS round trip ----
#pragma unroll
        for (int jf = 0; jf < 4; jf++)
#pragma unroll
            for (int r = 0; r < 8; r++)
                myP[(r + half * 8) * 72 + jf * 16 + lm] = (bf16_t)p[jf][r];
        __syncthreads();

        // ---- O += P @ V  (P A-frags + V B-frags from LDS) ----
#pragma unroll
        for (int ks = 0; ks < 2; ks++) {
            V16 ap;
            ap.h[0] = *(const v8bf*)(myP + lm * 72 + ks * 32 + klo);
            ap.h[1] = *(const v8bf*)(myP + lm * 72 + ks * 32 + klo + 8);
#pragma unroll
            for (int df = 0; df < 4; df++) {
                const bf16_t* vp = vt + (df * 16 + lm) * 72 + ks * 32 + kb;
                V16 bv;
                bv.v = *(const v16bf*)(vp);
                oacc[df] = __builtin_amdgcn_wmma_f32_16x16x32_bf16(
                    false, ap.v, false, bv.v, (short)0, oacc[df], false, false);
            }
        }
        // next iteration's top barrier separates these reads from buffer overwrite
    }

    // ---- normalize and store O as bf16 [B*S, E] (heads concatenated) ----
#pragma unroll
    for (int df = 0; df < 4; df++) {
#pragma unroll
        for (int r = 0; r < 8; r++) {
            const int s  = q0 + r + half * 8;
            const int gm = b * Ss + s;
            const int gn = h * Dd + df * 16 + lm;
            O[(size_t)gm * Ee + gn] = (bf16_t)(oacc[df][r] / rsum[r]);
        }
    }
}

extern "C" void kernel_launch(void* const* d_in, const int* in_sizes, int n_in,
                              void* d_out, int out_size, void* d_ws, size_t ws_size,
                              hipStream_t stream) {
    (void)in_sizes; (void)n_in; (void)out_size; (void)ws_size;
    const float* query = (const float*)d_in[0];
    const float* key_  = (const float*)d_in[1];
    const float* value = (const float*)d_in[2];
    const float* Wq = (const float*)d_in[3];
    const float* bq = (const float*)d_in[4];
    const float* Wk = (const float*)d_in[5];
    const float* bk = (const float*)d_in[6];
    const float* Wv = (const float*)d_in[7];
    const float* bv = (const float*)d_in[8];
    const float* Wo = (const float*)d_in[9];
    const float* bo = (const float*)d_in[10];
    float* out = (float*)d_out;

    char* ws = (char*)d_ws;
    const size_t MB = 1024 * 1024;
    bf16_t* xq = (bf16_t*)(ws + 0 * MB);
    bf16_t* xk = (bf16_t*)(ws + 8 * MB);
    bf16_t* xv = (bf16_t*)(ws + 16 * MB);
    bf16_t* wq = (bf16_t*)(ws + 24 * MB);
    bf16_t* wk = (bf16_t*)(ws + 26 * MB);
    bf16_t* wv = (bf16_t*)(ws + 28 * MB);
    bf16_t* wo = (bf16_t*)(ws + 30 * MB);
    bf16_t* Qh = (bf16_t*)(ws + 32 * MB);
    bf16_t* Kh = (bf16_t*)(ws + 40 * MB);
    bf16_t* Vt = (bf16_t*)(ws + 48 * MB);
    bf16_t* O  = (bf16_t*)(ws + 56 * MB);

    const int nAct = Mtot * Ee;  // 4M
    const int nW   = Ee * Ee;    // 1M
    const int TPB = 256;
    cvt_f32_bf16<<<(nAct + TPB - 1) / TPB, TPB, 0, stream>>>(query, xq, nAct);
    cvt_f32_bf16<<<(nAct + TPB - 1) / TPB, TPB, 0, stream>>>(key_,  xk, nAct);
    cvt_f32_bf16<<<(nAct + TPB - 1) / TPB, TPB, 0, stream>>>(value, xv, nAct);
    cvt_f32_bf16<<<(nW + TPB - 1) / TPB, TPB, 0, stream>>>(Wq, wq, nW);
    cvt_f32_bf16<<<(nW + TPB - 1) / TPB, TPB, 0, stream>>>(Wk, wk, nW);
    cvt_f32_bf16<<<(nW + TPB - 1) / TPB, TPB, 0, stream>>>(Wv, wv, nW);
    cvt_f32_bf16<<<(nW + TPB - 1) / TPB, TPB, 0, stream>>>(Wo, wo, nW);

    dim3 gg(Mtot / 128, Ee / 128);
    gemm_nt_bf16<0><<<gg, 256, 0, stream>>>(xq, wq, bq, Qh, nullptr, Mtot, Ee, Ee);
    gemm_nt_bf16<0><<<gg, 256, 0, stream>>>(xk, wk, bk, Kh, nullptr, Mtot, Ee, Ee);
    gemm_nt_bf16<1><<<gg, 256, 0, stream>>>(xv, wv, bv, Vt, nullptr, Mtot, Ee, Ee);

    dim3 ga(Ss / 64, Hh, Bb);
    flash_dft_attn<<<ga, 128, 0, stream>>>(Qh, Kh, Vt, O);

    gemm_nt_bf16<2><<<gg, 256, 0, stream>>>(O, wo, bo, nullptr, out, Mtot, Ee, Ee);
}
